// KSSMLayer_66872640798755
// MI455X (gfx1250) — matrix operand order repaired and checked by reading, not compiled, not run on hardware
//
#include <hip/hip_runtime.h>
#include <hip/hip_bf16.h>

// ---------------------------------------------------------------------------
// KSSM forward for MI455X (gfx1250), wave32 + v_wmma_f32_16x16x32_bf16.
// Operands pre-packed ONCE into WMMA-native tiled split-bf16 (hi+lo) layout,
// so GEMM inner loops are pure {global_load_b128 x N + v_wmma} with no
// per-iteration conversion VALU.  Complex chunked scan for the recurrence.
// ---------------------------------------------------------------------------

#define BB      2
#define SS      2048
#define DMODEL  512
#define DINNER  1024
#define MROWS   (BB * SS)       // 4096
#define NCHUNK  64
#define CLEN    (SS / NCHUNK)   // 32

typedef __attribute__((ext_vector_type(16))) __bf16 v16bf;
typedef __attribute__((ext_vector_type(8)))  float  v8f;

// ---- workspace layout (bytes), all 256B aligned ---------------------------
#define OFF_CBUF   0ull                       // f32 [MROWS*DINNER*2]   33.5MB
#define OFF_UBUF   33554432ull                // f32 [MROWS*DINNER*2]   33.5MB
#define OFF_CARRY  67108864ull                // f32 [BB*DINNER*NCHUNK*2]  1MB
#define OFF_XPACK  68157440ull                // bf16 tiles of x           8MB
#define OFF_WPACK  76546048ull                // bf16 tiles of 6 gate Ws  12MB
#define OFF_WOPACK 89128960ull                // bf16 tiles of W_out       4MB
// packed states reuse the (dead after scan) cBuf region: exactly 33.5MB

// ---- helpers --------------------------------------------------------------

__device__ __forceinline__ float softplus_f(float z) {
    return (z > 20.0f) ? z : log1pf(expf(z));
}

__device__ __forceinline__ void split2(float a0, float a1,
                                       v16bf& hi, v16bf& lo, int v) {
    __bf16 h0 = (__bf16)a0;
    __bf16 h1 = (__bf16)a1;
    hi[2 * v]     = h0;
    hi[2 * v + 1] = h1;
    lo[2 * v]     = (__bf16)(a0 - (float)h0);
    lo[2 * v + 1] = (__bf16)(a1 - (float)h1);
}

__device__ __forceinline__ v8f wmma3(const v16bf& ahi, const v16bf& alo,
                                     const v16bf& bhi, const v16bf& blo, v8f acc) {
    acc = __builtin_amdgcn_wmma_f32_16x16x32_bf16(false, ahi, false, bhi,
                                                  (short)0, acc, false, false);
    acc = __builtin_amdgcn_wmma_f32_16x16x32_bf16(false, ahi, false, blo,
                                                  (short)0, acc, false, false);
    acc = __builtin_amdgcn_wmma_f32_16x16x32_bf16(false, alo, false, bhi,
                                                  (short)0, acc, false, false);
    return acc;
}

// ---- pack kernels: f32 -> WMMA-native tiled split-bf16 --------------------
// Tile storage: 64 v16bf per tile = [32 lanes hi][32 lanes lo], 2KB.

// A operand (16x32 tiles, rows of src). tile id = rt*nKT + kt.
// 16-bit A layout: lane={m,half}; VGPR v: K = half*16 + 2v, 2v+1
__global__ __launch_bounds__(256)
void pack_a(const float* __restrict__ src, int lda, int nKT,
            v16bf* __restrict__ dst) {
    int flat = blockIdx.x * 256 + threadIdx.x;
    int lane = flat & 31;
    int tile = flat >> 5;
    int kt = tile % nKT;
    int rt = tile / nKT;
    int m16 = lane & 15, half = lane >> 4;
    const float* arow = src + (size_t)(rt * 16 + m16) * lda + kt * 32;
    v16bf hi, lo;
#pragma unroll
    for (int v = 0; v < 8; ++v) {
        int k = half * 16 + (v << 1);
        split2(arow[k], arow[k + 1], hi, lo, v);
    }
    size_t tb = (size_t)tile * 64;
    dst[tb + lane]      = hi;
    dst[tb + 32 + lane] = lo;
}

// B operand (32x16 tiles, W row-major W[k*ldw+col]). tile id = nt*nKT + kt.
// 16-bit B layout used here (contiguous k, not ISA order): lane={n,half}; VGPR v: K = half*16 + 2v, 2v+1
__global__ __launch_bounds__(256)
void pack_b(const float* __restrict__ src, int ldw, int cmul, int cadd, int nKT,
            v16bf* __restrict__ dst) {
    int flat = blockIdx.x * 256 + threadIdx.x;
    int lane = flat & 31;
    int tile = flat >> 5;
    int kt = tile % nKT;
    int nt = tile / nKT;
    int n = lane & 15, half = lane >> 4;
    int col = (nt * 16 + n) * cmul + cadd;
    const float* wcol = src + (size_t)(kt * 32 + half * 16) * ldw + col;
    v16bf hi, lo;
#pragma unroll
    for (int v = 0; v < 8; ++v) {
        float w0 = wcol[(size_t)(2 * v) * ldw];
        float w1 = wcol[(size_t)(2 * v + 1) * ldw];
        split2(w0, w1, hi, lo, v);
    }
    size_t tb = (size_t)tile * 64;
    dst[tb + lane]      = hi;
    dst[tb + 32 + lane] = lo;
}

// ---- kernel 1: fused gate GEMMs + elementwise epilogue --------------------
// grid (MROWS/16, DINNER/16), block 32 (one wave, EXEC all ones for WMMA)
__global__ __launch_bounds__(32)
void kssm_gates(const v16bf* __restrict__ Ap, const v16bf* __restrict__ Bp,
                float* __restrict__ cBuf, float* __restrict__ uBuf) {
    const int lane = threadIdx.x;
    const int m16  = lane & 15;
    const int half = lane >> 4;
    const int rt = blockIdx.x;
    const int nt = blockIdx.y;

    v8f acc[6] = {};   // xin, alpha, omega, dt, B0, B1
    const int nKT = DMODEL / 32;   // 16

    for (int kt = 0; kt < nKT; ++kt) {
        size_t at = ((size_t)rt * nKT + kt) * 64;
        v16bf ahi = Ap[at + lane];
        v16bf alo = Ap[at + 32 + lane];
        if (kt + 1 < nKT)
            __builtin_prefetch(&Ap[at + 64 + lane], 0, 1);
#pragma unroll
        for (int o = 0; o < 6; ++o) {
            size_t bt = ((size_t)o * 64 * nKT + (size_t)nt * nKT + kt) * 64;
            v16bf bhi = Bp[bt + lane];
            v16bf blo = Bp[bt + 32 + lane];
            acc[o] = wmma3(ahi, alo, bhi, blo, acc[o]);
        }
    }

    // Epilogue: C-layout VGPR r -> M = r + half*8, N = m16
#pragma unroll
    for (int r = 0; r < 8; ++r) {
        const int row = rt * 16 + r + half * 8;   // = b*SS + s
        const int col = nt * 16 + m16;            // channel j

        float xin = acc[0][r];
        float alp = softplus_f(acc[1][r]);
        float om  = acc[2][r];
        float dt  = softplus_f(acc[3][r]);
        float b0  = acc[4][r];
        float b1  = acc[5][r];

        float a     = 0.5f * dt * alp;
        float w     = 0.5f * dt * om;
        float onepa = 1.0f + a;
        float D     = onepa * onepa + w * w + 1e-6f;
        float invD  = 1.0f / D;

        float c_re = (1.0f - a * a - w * w) * invD;   // a_diag
        float c_im = -(2.0f * w) * invD;              // -a_off (complex form)

        float u1 = dt * b0 * xin;
        float u2 = dt * b1 * xin;
        float ub1 = (onepa * u1 + w * u2) * invD;
        float ub2 = (onepa * u2 - w * u1) * invD;

        size_t idx = ((size_t)row * DINNER + col) * 2;
        *reinterpret_cast<float2*>(cBuf + idx) = make_float2(c_re, c_im);
        *reinterpret_cast<float2*>(uBuf + idx) = make_float2(ub1, ub2);
    }
}

// ---- kernel 2a: in-place local scan per (channel, chunk) ------------------
__global__ __launch_bounds__(256)
void kssm_scan_local(float* __restrict__ cBuf, float* __restrict__ uBuf) {
    const int ch    = blockIdx.y * blockDim.x + threadIdx.x;  // 0..2047
    const int chunk = blockIdx.x;
    const int b = ch >> 10;
    const int j = ch & (DINNER - 1);
    size_t base = ((size_t)(b * SS + chunk * CLEN) * DINNER + j) * 2;

    float pr = 1.0f, pi_ = 0.0f;
    float vr = 0.0f, vi  = 0.0f;
    for (int t = 0; t < CLEN; ++t) {
        size_t idx = base + (size_t)t * (DINNER * 2);
        float2 c = *reinterpret_cast<const float2*>(cBuf + idx);
        float2 u = *reinterpret_cast<const float2*>(uBuf + idx);
        float nvr = c.x * vr - c.y * vi + u.x;
        float nvi = c.x * vi + c.y * vr + u.y;
        float npr = c.x * pr - c.y * pi_;
        float npi = c.x * pi_ + c.y * pr;
        vr = nvr; vi = nvi; pr = npr; pi_ = npi;
        *reinterpret_cast<float2*>(cBuf + idx) = make_float2(pr, pi_);
        *reinterpret_cast<float2*>(uBuf + idx) = make_float2(vr, vi);
    }
}

// ---- kernel 2b: carry scan across chunks + final_state --------------------
__global__ __launch_bounds__(256)
void kssm_scan_carry(const float* __restrict__ cBuf, const float* __restrict__ uBuf,
                     const float* __restrict__ st0,
                     float* __restrict__ carry, float* __restrict__ finalOut) {
    const int ch = blockIdx.x * blockDim.x + threadIdx.x;   // = b*DINNER + j
    const int b = ch >> 10;
    const int j = ch & (DINNER - 1);

    float hr  = st0[(size_t)ch * 2];
    float hi_ = st0[(size_t)ch * 2 + 1];
    for (int k = 0; k < NCHUNK; ++k) {
        reinterpret_cast<float2*>(carry)[(size_t)k * (BB * DINNER) + ch] =
            make_float2(hr, hi_);
        size_t idx = ((size_t)(b * SS + k * CLEN + (CLEN - 1)) * DINNER + j) * 2;
        float2 P = *reinterpret_cast<const float2*>(cBuf + idx);
        float2 V = *reinterpret_cast<const float2*>(uBuf + idx);
        float nr = P.x * hr - P.y * hi_ + V.x;
        float ni = P.x * hi_ + P.y * hr + V.y;
        hr = nr; hi_ = ni;
    }
    finalOut[(size_t)ch * 2]     = hr;
    finalOut[(size_t)ch * 2 + 1] = hi_;
}

// ---- kernel 2c: apply chunk prefix, states overwrite uBuf in place --------
__global__ __launch_bounds__(256)
void kssm_scan_apply(const float* __restrict__ cBuf, float* __restrict__ uBuf,
                     const float* __restrict__ carry) {
    const int ch    = blockIdx.y * blockDim.x + threadIdx.x;
    const int chunk = blockIdx.x;
    const int b = ch >> 10;
    const int j = ch & (DINNER - 1);
    float2 h = reinterpret_cast<const float2*>(carry)[(size_t)chunk * (BB * DINNER) + ch];
    size_t base = ((size_t)(b * SS + chunk * CLEN) * DINNER + j) * 2;
    for (int t = 0; t < CLEN; ++t) {
        size_t idx = base + (size_t)t * (DINNER * 2);
        float2 p = *reinterpret_cast<const float2*>(cBuf + idx);
        float2 v = *reinterpret_cast<const float2*>(uBuf + idx);
        float sr = p.x * h.x - p.y * h.y + v.x;
        float si = p.x * h.y + p.y * h.x + v.y;
        *reinterpret_cast<float2*>(uBuf + idx) = make_float2(sr, si);
    }
}

// ---- kernel 3: output GEMM  states(4096x2048) @ W_out(2048x512) -----------
// grid (MROWS/16, DMODEL/16), block 32
__global__ __launch_bounds__(32)
void kssm_out_gemm(const v16bf* __restrict__ Sp, const v16bf* __restrict__ Wp,
                   float* __restrict__ Out) {
    const int lane = threadIdx.x;
    const int m16  = lane & 15;
    const int half = lane >> 4;
    const int rt = blockIdx.x;
    const int nt = blockIdx.y;
    const int nKT = (2 * DINNER) / 32;   // 64

    v8f acc = {};
    for (int kt = 0; kt < nKT; ++kt) {
        size_t at = ((size_t)rt * nKT + kt) * 64;
        size_t bt = ((size_t)nt * nKT + kt) * 64;
        v16bf ahi = Sp[at + lane];
        v16bf alo = Sp[at + 32 + lane];
        v16bf bhi = Wp[bt + lane];
        v16bf blo = Wp[bt + 32 + lane];
        if (kt + 1 < nKT)
            __builtin_prefetch(&Sp[at + 64 + lane], 0, 1);
        acc = wmma3(ahi, alo, bhi, blo, acc);
    }

#pragma unroll
    for (int r = 0; r < 8; ++r) {
        Out[(size_t)(rt * 16 + r + half * 8) * DMODEL + nt * 16 + m16] = acc[r];
    }
}

// ---------------------------------------------------------------------------
extern "C" void kernel_launch(void* const* d_in, const int* in_sizes, int n_in,
                              void* d_out, int out_size, void* d_ws, size_t ws_size,
                              hipStream_t stream) {
    (void)in_sizes; (void)n_in; (void)out_size; (void)ws_size;

    const float* x    = (const float*)d_in[0];
    const float* st0  = (const float*)d_in[1];
    const float* Wx   = (const float*)d_in[2];
    const float* Wa   = (const float*)d_in[3];
    const float* Wo   = (const float*)d_in[4];
    const float* Wd   = (const float*)d_in[5];
    const float* Wb   = (const float*)d_in[6];
    const float* Wout = (const float*)d_in[7];

    float* out      = (float*)d_out;
    float* finalOut = out + (size_t)MROWS * DMODEL;

    char* ws = (char*)d_ws;
    float*  cBuf   = (float*)(ws + OFF_CBUF);
    float*  uBuf   = (float*)(ws + OFF_UBUF);
    float*  carry  = (float*)(ws + OFF_CARRY);
    v16bf*  xpack  = (v16bf*)(ws + OFF_XPACK);
    v16bf*  wpack  = (v16bf*)(ws + OFF_WPACK);
    v16bf*  wopack = (v16bf*)(ws + OFF_WOPACK);
    v16bf*  spack  = (v16bf*)(ws + OFF_CBUF);   // reuse cBuf after scan

    // --- pack x and weights into WMMA-native split-bf16 tiles ---
    // x: 256 row-tiles x 16 k-tiles
    pack_a<<<(256 * 16 * 32) / 256, 256, 0, stream>>>(x, DMODEL, 16, xpack);
    // 6 gate matrices: each 64 n-tiles x 16 k-tiles (1024 tiles = 64 v16bf each)
    const size_t WT = 64 * 16 * 64;   // v16bf per gate matrix
    pack_b<<<(64 * 16 * 32) / 256, 256, 0, stream>>>(Wx, DINNER, 1, 0, 16, wpack + 0 * WT);
    pack_b<<<(64 * 16 * 32) / 256, 256, 0, stream>>>(Wa, DINNER, 1, 0, 16, wpack + 1 * WT);
    pack_b<<<(64 * 16 * 32) / 256, 256, 0, stream>>>(Wo, DINNER, 1, 0, 16, wpack + 2 * WT);
    pack_b<<<(64 * 16 * 32) / 256, 256, 0, stream>>>(Wd, DINNER, 1, 0, 16, wpack + 3 * WT);
    pack_b<<<(64 * 16 * 32) / 256, 256, 0, stream>>>(Wb, 2 * DINNER, 2, 0, 16, wpack + 4 * WT);
    pack_b<<<(64 * 16 * 32) / 256, 256, 0, stream>>>(Wb, 2 * DINNER, 2, 1, 16, wpack + 5 * WT);
    // W_out: 32 n-tiles x 64 k-tiles
    pack_b<<<(32 * 64 * 32) / 256, 256, 0, stream>>>(Wout, DMODEL, 1, 0, 64, wopack);

    // --- gate GEMMs + epilogue -> c (complex coeff), u_bar ---
    kssm_gates<<<dim3(MROWS / 16, DINNER / 16), 32, 0, stream>>>(
        xpack, wpack, cBuf, uBuf);

    // --- chunked complex scan ---
    kssm_scan_local<<<dim3(NCHUNK, (BB * DINNER) / 256), 256, 0, stream>>>(cBuf, uBuf);
    kssm_scan_carry<<<dim3((BB * DINNER) / 256), 256, 0, stream>>>(
        cBuf, uBuf, st0, carry, finalOut);
    kssm_scan_apply<<<dim3(NCHUNK, (BB * DINNER) / 256), 256, 0, stream>>>(
        cBuf, uBuf, carry);

    // --- pack states (uBuf f32 -> spack tiles in dead cBuf region) ---
    // 256 row-tiles x 64 k-tiles over (4096 x 2048)
    pack_a<<<(256 * 64 * 32) / 256, 256, 0, stream>>>(uBuf, 2 * DINNER, 64, spack);

    // --- output GEMM ---
    kssm_out_gemm<<<dim3(MROWS / 16, DMODEL / 16), 32, 0, stream>>>(spack, wopack, out);
}
